// DecoderBlock_28802050687383
// MI455X (gfx1250) — compile-verified, matrix operand order checked by reading, not run on hardware
//
#include <hip/hip_runtime.h>
#include <hip/hip_bf16.h>

// ---------------------------------------------------------------------------
// MI455X (gfx1250) decoder block.
//   ~275 GFLOP total vs ~100MB traffic @ 23.3 TB/s  -> matrix-core bound.
//   All GEMMs + attention einsums via v_wmma_f32_16x16x32_f16 (wave32),
//   fp32 accumulate, layernorm/softmax in fp32 VALU.
//   GEMM tiles staged with the Tensor Data Mover (tensor_load_to_lds,
//   TENSORcnt-tracked, double-buffered) -- zero VGPR staging traffic.
// ---------------------------------------------------------------------------

typedef __attribute__((ext_vector_type(16))) _Float16     v16h;
typedef __attribute__((ext_vector_type(8)))  float        v8f;
typedef __attribute__((ext_vector_type(4)))  unsigned int su4;
typedef __attribute__((ext_vector_type(8)))  unsigned int su8;

#define WMMA_F16(a, b, c) \
  __builtin_amdgcn_wmma_f32_16x16x32_f16(false, (a), false, (b), (short)0, (c), false, false)

union FragH { v16h v; unsigned int u[8]; };

// Row-major 16x32 f16 fragment load matching ISA 7.12.2 A/B layout:
//   lanes 0-15 : row = lane,    K = 0..7  (VGPR0-3) and 16..23 (VGPR4-7)
//   lanes 16-31: row = lane-16, K = 8..15 (VGPR0-3) and 24..31 (VGPR4-7)
__device__ inline v16h load_frag_rm(const _Float16* p0, int ld, int lane) {
  int row = lane & 15;
  int kb  = (lane >> 4) << 3;   // 0 or 8
  const _Float16* p = p0 + (size_t)row * ld + kb;
  FragH f;
#pragma unroll
  for (int i = 0; i < 4; ++i) {
    f.u[i]     = *(const unsigned int*)(p + 2 * i);
    f.u[i + 4] = *(const unsigned int*)(p + 16 + 2 * i);
  }
  return f.v;
}

// ---------------------------------------------------------------------------
// TDM: DMA a 128-row x 32-half f16 tile (row stride K halves) global -> LDS.
// 2D descriptor (groups 0+1 only, VADDR2/3 = NULL). Pad feature inserts
// 4 dwords after every 16 dwords -> LDS row stride 40 halves (== LDT).
// Wave-level op: EXEC ignored, tracked by TENSORcnt. (ISA 08_async_tensor §8)
// ---------------------------------------------------------------------------
__device__ inline void tdm_load_tile(unsigned lds_off, const _Float16* g, int K) {
  unsigned long long ga = (unsigned long long)(const void*)g;
  su4 g0;
  g0[0] = 1u;                                   // count=1, user mode, no gather
  g0[1] = lds_off;                              // LDS byte address
  g0[2] = (unsigned)ga;                         // global_addr[31:0]
  g0[3] = ((unsigned)(ga >> 32) & 0x1FFFFFFu)   // global_addr[56:32]
          | (2u << 30);                         // type = 2 ("image")
  su8 g1;
  g1[0] = (1u << 16)                            // data_size = 2 bytes
        | (1u << 20)                            // pad_enable
        | (3u << 22)                            // pad_interval: every 16 dwords
        | (3u << 25);                           // pad_amount:   4 dwords
  unsigned d0 = (unsigned)K;                    // tensor_dim0 (halves)
  unsigned d1 = 0x40000u;                       // tensor_dim1 (large, no OOB)
  g1[1] = (d0 & 0xFFFFu) << 16;                 // tensor_dim0[15:0]
  g1[2] = (d0 >> 16) | ((d1 & 0xFFFFu) << 16);  // tensor_dim0[31:16], dim1[15:0]
  g1[3] = (d1 >> 16) | (32u << 16);             // dim1[31:16], tile_dim0 = 32
  g1[4] = 128u;                                 // tile_dim1 = 128, tile_dim2 = 0
  g1[5] = (unsigned)K;                          // tensor_dim0_stride[31:0]
  g1[6] = 0u;                                   // stride hi, dim1_stride lo
  g1[7] = 0u;
  asm volatile("tensor_load_to_lds %0, %1" :: "s"(g0), "s"(g1) : "memory");
}

// ---------------------------------------------------------------------------
// Weight cast + transpose: WT[n*K + k] = (f16) W[k*N + n]
// ---------------------------------------------------------------------------
__global__ void castT_kernel(const float* __restrict__ W, _Float16* __restrict__ WT,
                             int K, int N) {
  size_t idx = (size_t)blockIdx.x * blockDim.x + threadIdx.x;
  size_t total = (size_t)K * N;
  if (idx >= total) return;
  int n = (int)(idx / K);
  int k = (int)(idx - (size_t)n * K);
  WT[idx] = (_Float16)W[(size_t)k * N + n];
}

// ---------------------------------------------------------------------------
// LayerNorm (fp32 in) -> f16 out. One block per token row. C = 1024.
// ---------------------------------------------------------------------------
__global__ __launch_bounds__(128)
void ln_cast_kernel(const float* __restrict__ x, const float* __restrict__ w,
                    const float* __restrict__ b, _Float16* __restrict__ out, int C) {
  int row = blockIdx.x;
  const float* xr = x + (size_t)row * C;
  float sum = 0.f, sq = 0.f;
  for (int c = threadIdx.x; c < C; c += blockDim.x) {
    float v = xr[c]; sum += v; sq += v * v;
  }
#pragma unroll
  for (int m = 16; m >= 1; m >>= 1) {
    sum += __shfl_xor(sum, m, 32);
    sq  += __shfl_xor(sq,  m, 32);
  }
  __shared__ float s_sum[8], s_sq[8];
  int wid = threadIdx.x >> 5;
  if ((threadIdx.x & 31) == 0) { s_sum[wid] = sum; s_sq[wid] = sq; }
  __syncthreads();
  float tot = 0.f, totsq = 0.f;
  int nw = blockDim.x >> 5;
  for (int i = 0; i < nw; ++i) { tot += s_sum[i]; totsq += s_sq[i]; }
  float mu   = tot / C;
  float var  = totsq / C - mu * mu;
  float rstd = rsqrtf(var + 1e-5f);
  for (int c = threadIdx.x; c < C; c += blockDim.x) {
    float v = (xr[c] - mu) * rstd * w[c] + b[c];
    out[(size_t)row * C + c] = (_Float16)v;
  }
}

// ---------------------------------------------------------------------------
// WMMA GEMM: C[M,N] = A[M,K] @ B[K,N] + bias,  A f16 row-major, B as BT[N,K].
// Block tile 128x128, BK=32, 8 waves, wave tile 64x32 (8 wmma acc).
// Double-buffered LDS; tiles DMA'd by TDM from wave 0, one tile ahead.
// MODE 0: split-scatter to q/k f16 [B,H,T,64] and vT f16 [B,H,64,T]
// MODE 1: outF = resid + acc + bias  (fp32)
// MODE 2: outH = relu(acc + bias)    (f16)
// ---------------------------------------------------------------------------
#define BM 128
#define BN 128
#define BK 32
#define LDT 40   // padded LDS stride in halves (from TDM pad: 16dw + 4dw pad)

template<int MODE>
__global__ __launch_bounds__(256)
void gemm_wmma(const _Float16* __restrict__ A, const _Float16* __restrict__ BT,
               const float* __restrict__ bias, int M, int N, int K,
               const float* __restrict__ resid, float* __restrict__ outF,
               _Float16* __restrict__ outH,
               _Float16* __restrict__ qb, _Float16* __restrict__ kb,
               _Float16* __restrict__ vT, int T) {
  __shared__ __align__(16) _Float16 sA[2][BM * LDT];
  __shared__ __align__(16) _Float16 sB[2][BN * LDT];
  int tid   = threadIdx.x;
  int lane  = tid & 31;
  int wid   = tid >> 5;
  int waveM = wid >> 2;  // 0..1  -> 64-row slab
  int waveN = wid & 3;   // 0..3  -> 32-col slab
  int m0 = blockIdx.y * BM;
  int n0 = blockIdx.x * BN;

  const _Float16* gA = A  + (size_t)m0 * K;   // tile origin, k0 advances
  const _Float16* gB = BT + (size_t)n0 * K;
  unsigned ldsA[2], ldsB[2];
#pragma unroll
  for (int i = 0; i < 2; ++i) {
    ldsA[i] = (unsigned)(unsigned long long)(void*)&sA[i][0];
    ldsB[i] = (unsigned)(unsigned long long)(void*)&sB[i][0];
  }

  v8f acc[4][2];
#pragma unroll
  for (int mi = 0; mi < 4; ++mi)
#pragma unroll
    for (int ni = 0; ni < 2; ++ni) acc[mi][ni] = {};

  // Prologue: DMA first tile pair into buffer 0
  if (wid == 0) {
    tdm_load_tile(ldsA[0], gA, K);
    tdm_load_tile(ldsB[0], gB, K);
  }

  int cur = 0;
  for (int k0 = 0; k0 < K; k0 += BK) {
    int nxt = cur ^ 1;
    if (wid == 0) {
      if (k0 + BK < K) {
        // Issue next tile pair, then wait for the current pair (in-order).
        tdm_load_tile(ldsA[nxt], gA + k0 + BK, K);
        tdm_load_tile(ldsB[nxt], gB + k0 + BK, K);
        __builtin_amdgcn_s_wait_tensorcnt(2);
      } else {
        __builtin_amdgcn_s_wait_tensorcnt(0);
      }
    }
    __syncthreads();   // current-tile DMA complete & visible to all waves

    v16h af[4], bf[2];
#pragma unroll
    for (int mi = 0; mi < 4; ++mi)
      af[mi] = load_frag_rm(&sA[cur][(waveM * 64 + mi * 16) * LDT], LDT, lane);
#pragma unroll
    for (int ni = 0; ni < 2; ++ni)
      bf[ni] = load_frag_rm(&sB[cur][(waveN * 32 + ni * 16) * LDT], LDT, lane);
#pragma unroll
    for (int mi = 0; mi < 4; ++mi)
#pragma unroll
      for (int ni = 0; ni < 2; ++ni)
        acc[mi][ni] = WMMA_F16(af[mi], bf[ni], acc[mi][ni]);

    __syncthreads();   // all reads of buffer `cur` done before it is reused
    cur = nxt;
  }

  // Epilogue (ISA C-layout: lane = col-in-tile, VGPR r = row; lanes>=16 -> +8)
  int rbase = (lane >> 4) << 3;
  int nlane = lane & 15;
#pragma unroll
  for (int mi = 0; mi < 4; ++mi) {
#pragma unroll
    for (int ni = 0; ni < 2; ++ni) {
#pragma unroll
      for (int r = 0; r < 8; ++r) {
        int gm = m0 + waveM * 64 + mi * 16 + rbase + r;
        int gn = n0 + waveN * 32 + ni * 16 + nlane;
        float v = acc[mi][ni][r] + bias[gn];
        if (MODE == 0) {
          int bidx = gm / T, t = gm % T;
          int sec = gn >> 10, cc = gn & 1023, hh = cc >> 6, dd = cc & 63;
          size_t bh = (size_t)bidx * 16 + hh;
          _Float16 hv = (_Float16)v;
          if (sec == 0)      qb[(bh * T + t) * 64 + dd] = hv;
          else if (sec == 1) kb[(bh * T + t) * 64 + dd] = hv;
          else               vT[(bh * 64 + dd) * T + t] = hv;
        } else if (MODE == 1) {
          outF[(size_t)gm * N + gn] = resid[(size_t)gm * N + gn] + v;
        } else {
          outH[(size_t)gm * N + gn] = (_Float16)fmaxf(v, 0.f);
        }
      }
    }
  }
}

// ---------------------------------------------------------------------------
// Flash attention, causal, wave32. One wave = 16 query rows of one (b,h).
// S = q @ k^T (WMMA, K=64), online softmax (f32), O += P @ v (WMMA).
// P re-layout (C-layout -> A-fragment) bounced through per-wave LDS tile.
// ---------------------------------------------------------------------------
__global__ __launch_bounds__(256)
void attn_kernel(const _Float16* __restrict__ q, const _Float16* __restrict__ kbuf,
                 const _Float16* __restrict__ vT, _Float16* __restrict__ y,
                 int T, int H, int HD) {
  __shared__ __align__(16) _Float16 sP[8][16 * LDT];  // per-wave 16x32 P tile
  int lane = threadIdx.x & 31;
  int wid  = threadIdx.x >> 5;
  int nQT  = T >> 4;                       // 128 query tiles per (b,h)
  int g    = blockIdx.x * 8 + wid;
  int qt   = g & (nQT - 1);
  int bh   = g / nQT;
  int i0   = qt << 4;
  int C    = H * HD;

  const _Float16* qbase = q + ((size_t)bh * T + i0) * HD;
  v16h qa0 = load_frag_rm(qbase,      HD, lane);   // K = 0..31
  v16h qa1 = load_frag_rm(qbase + 32, HD, lane);   // K = 32..63

  v8f o[4];
#pragma unroll
  for (int dt = 0; dt < 4; ++dt) o[dt] = {};
  float mrun[8], lrun[8];
#pragma unroll
  for (int r = 0; r < 8; ++r) { mrun[r] = -1e30f; lrun[r] = 0.f; }

  const float scale = 0.03125f;            // 1/sqrt(C) = 1/32
  int rbase = (lane >> 4) << 3;
  int nl    = lane & 15;
  _Float16* myP = &sP[wid][0];
  int kblocks = (i0 >> 5) + 1;             // causal: only blocks with j_min <= i_max

  for (int kb = 0; kb < kblocks; ++kb) {
    int j0 = kb << 5;
    if (kb + 1 < kblocks) {                // prefetch next key block
      __builtin_prefetch(kbuf + ((size_t)bh * T + j0 + 32) * HD, 0, 0);
      __builtin_prefetch(vT + (size_t)bh * HD * T + j0 + 32, 0, 0);
    }
    v8f s[2]; s[0] = {}; s[1] = {};
#pragma unroll
    for (int nt = 0; nt < 2; ++nt) {
      const _Float16* kt = kbuf + ((size_t)bh * T + j0 + nt * 16) * HD;
      v16h b0 = load_frag_rm(kt,      HD, lane);
      v16h b1 = load_frag_rm(kt + 32, HD, lane);
      s[nt] = WMMA_F16(qa0, b0, s[nt]);
      s[nt] = WMMA_F16(qa1, b1, s[nt]);
    }
    // Online softmax update per row (rows striped across VGPRs / lane halves)
#pragma unroll
    for (int r = 0; r < 8; ++r) {
      int i = i0 + rbase + r;
      float v0 = s[0][r] * scale; if (j0 + nl      > i) v0 = -1e30f;
      float v1 = s[1][r] * scale; if (j0 + 16 + nl > i) v1 = -1e30f;
      float mx = fmaxf(v0, v1);
#pragma unroll
      for (int m = 8; m >= 1; m >>= 1) mx = fmaxf(mx, __shfl_xor(mx, m, 32));
      float mnew = fmaxf(mrun[r], mx);
      float cor  = __expf(mrun[r] - mnew);
      float p0   = __expf(v0 - mnew);
      float p1   = __expf(v1 - mnew);
      float ps   = p0 + p1;
#pragma unroll
      for (int m = 8; m >= 1; m >>= 1) ps += __shfl_xor(ps, m, 32);
      lrun[r] = lrun[r] * cor + ps;
      mrun[r] = mnew;
#pragma unroll
      for (int dt = 0; dt < 4; ++dt) o[dt][r] *= cor;
      myP[(rbase + r) * LDT + nl]      = (_Float16)p0;
      myP[(rbase + r) * LDT + 16 + nl] = (_Float16)p1;
    }
    asm volatile("s_wait_dscnt 0" ::: "memory");   // P tile visible to own wave
    v16h pf = load_frag_rm(myP, LDT, lane);        // A-fragment of P (16x32)
#pragma unroll
    for (int dt = 0; dt < 4; ++dt) {
      const _Float16* vb = vT + ((size_t)bh * HD + dt * 16) * T + j0;
      v16h vf = load_frag_rm(vb, T, lane);         // B^T rows = v^T rows (d-major)
      o[dt] = WMMA_F16(pf, vf, o[dt]);
    }
  }

  // Normalize and scatter into y f16 [B,T,C] at head offset
  int b  = bh / H, hh = bh % H;
#pragma unroll
  for (int r = 0; r < 8; ++r) {
    float inv = 1.0f / lrun[r];
    int t = i0 + rbase + r;
    size_t rowoff = ((size_t)b * T + t) * C + (size_t)hh * HD;
#pragma unroll
    for (int dt = 0; dt < 4; ++dt)
      y[rowoff + dt * 16 + nl] = (_Float16)(o[dt][r] * inv);
  }
}

// ---------------------------------------------------------------------------
// Host-side orchestration
// ---------------------------------------------------------------------------
extern "C" void kernel_launch(void* const* d_in, const int* in_sizes, int n_in,
                              void* d_out, int out_size, void* d_ws, size_t ws_size,
                              hipStream_t stream) {
  const int Bb = 4, Tt = 2048, C = 1024, H = 16, HD = 64;
  const int M = Bb * Tt;            // 8192 token rows
  (void)in_sizes; (void)n_in; (void)out_size; (void)ws_size;

  const float* x      = (const float*)d_in[0];
  const float* ln1_w  = (const float*)d_in[1];
  const float* ln1_b  = (const float*)d_in[2];
  const float* qkv_w  = (const float*)d_in[3];
  const float* qkv_b  = (const float*)d_in[4];
  const float* proj_w = (const float*)d_in[5];
  const float* proj_b = (const float*)d_in[6];
  const float* ln2_w  = (const float*)d_in[7];
  const float* ln2_b  = (const float*)d_in[8];
  const float* ff_w1  = (const float*)d_in[9];
  const float* ff_b1  = (const float*)d_in[10];
  const float* ff_w2  = (const float*)d_in[11];
  const float* ff_b2  = (const float*)d_in[12];

  // Workspace carve-up (256B aligned slabs)
  char* p = (char*)d_ws;
  auto take = [&](size_t bytes) { char* r = p; p += (bytes + 255) & ~(size_t)255; return r; };
  _Float16* h1     = (_Float16*)take((size_t)M * C * 2);        // LN1 out
  _Float16* h2     = (_Float16*)take((size_t)M * C * 2);        // LN2 out
  _Float16* ybuf   = (_Float16*)take((size_t)M * C * 2);        // attn out
  _Float16* a1     = (_Float16*)take((size_t)M * 4 * C * 2);    // relu(ff1)
  _Float16* qbuf   = (_Float16*)take((size_t)M * C * 2);        // [B,H,T,64]
  _Float16* kbuf   = (_Float16*)take((size_t)M * C * 2);        // [B,H,T,64]
  _Float16* vTbuf  = (_Float16*)take((size_t)M * C * 2);        // [B,H,64,T]
  float*    x1     = (float*)take((size_t)M * C * 4);           // post-attn residual
  _Float16* WqkvT  = (_Float16*)take((size_t)3 * C * C * 2);
  _Float16* WprojT = (_Float16*)take((size_t)C * C * 2);
  _Float16* Wff1T  = (_Float16*)take((size_t)4 * C * C * 2);
  _Float16* Wff2T  = (_Float16*)take((size_t)4 * C * C * 2);

  auto cdiv = [](size_t a, size_t b) { return (unsigned)((a + b - 1) / b); };

  // 1) Cast + transpose weights to f16 [N][K]
  castT_kernel<<<cdiv((size_t)C * 3 * C, 256), 256, 0, stream>>>(qkv_w,  WqkvT,  C,     3 * C);
  castT_kernel<<<cdiv((size_t)C * C,     256), 256, 0, stream>>>(proj_w, WprojT, C,     C);
  castT_kernel<<<cdiv((size_t)C * 4 * C, 256), 256, 0, stream>>>(ff_w1,  Wff1T,  C,     4 * C);
  castT_kernel<<<cdiv((size_t)4 * C * C, 256), 256, 0, stream>>>(ff_w2,  Wff2T,  4 * C, C);

  // 2) LN1 -> h1 (f16)
  ln_cast_kernel<<<M, 128, 0, stream>>>(x, ln1_w, ln1_b, h1, C);

  // 3) QKV GEMM: [8192,1024] @ [1024,3072], epilogue scatters q/k/vT
  gemm_wmma<0><<<dim3(3 * C / BN, M / BM), 256, 0, stream>>>(
      h1, WqkvT, qkv_b, M, 3 * C, C, nullptr, nullptr, nullptr,
      qbuf, kbuf, vTbuf, Tt);

  // 4) Causal flash attention -> ybuf (f16), 8192 wave-tasks / 8 per block
  attn_kernel<<<(Bb * H * (Tt / 16)) / 8, 256, 0, stream>>>(
      qbuf, kbuf, vTbuf, ybuf, Tt, H, HD);

  // 5) proj GEMM + residual: x1 = x + y @ proj_w + proj_b   (fp32)
  gemm_wmma<1><<<dim3(C / BN, M / BM), 256, 0, stream>>>(
      ybuf, WprojT, proj_b, M, C, C, x, x1, nullptr,
      nullptr, nullptr, nullptr, Tt);

  // 6) LN2 -> h2 (f16)
  ln_cast_kernel<<<M, 128, 0, stream>>>(x1, ln2_w, ln2_b, h2, C);

  // 7) FF1 GEMM + ReLU: a1 = relu(h2 @ ff_w1 + ff_b1)  (f16)
  gemm_wmma<2><<<dim3(4 * C / BN, M / BM), 256, 0, stream>>>(
      h2, Wff1T, ff_b1, M, 4 * C, C, nullptr, nullptr, a1,
      nullptr, nullptr, nullptr, Tt);

  // 8) FF2 GEMM + residual: out = x1 + a1 @ ff_w2 + ff_b2  (fp32)
  gemm_wmma<1><<<dim3(C / BN, M / BM), 256, 0, stream>>>(
      a1, Wff2T, ff_b2, M, C, 4 * C, x1, (float*)d_out, nullptr,
      nullptr, nullptr, nullptr, Tt);
}
